// Trm_63574105915418
// MI455X (gfx1250) — matrix operand order repaired and checked by reading, not compiled, not run on hardware
//
#include <hip/hip_runtime.h>
#include <hip/hip_bf16.h>

typedef unsigned char  u8;
typedef unsigned short u16;
typedef unsigned int   u32;

typedef __attribute__((ext_vector_type(16))) __bf16 v16bf;
typedef __attribute__((ext_vector_type(8)))  float  v8f;

union Frag { v16bf f; u32 u[8]; };

#define kB   64
#define kS   512
#define kD   200
#define kDP  224      // D padded to multiple of 32 for WMMA K
#define kH   8
#define kHD  200      // head dim
#define kDH  1600     // H * head dim
#define kL   6
#define kC   10
#define kM   (kB * kS)   // 32768 rows

static __device__ __forceinline__ u16 f2bf(float f) {
    u32 u = __float_as_uint(f);
    u += 0x7FFFu + ((u >> 16) & 1u);   // round-to-nearest-even
    return (u16)(u >> 16);
}

// Pack two f32 -> two bf16 in one dword (avoids 16-bit half-reg moves)
static __device__ __forceinline__ u32 pack2bf(float lo, float hi) {
    u32 ul = __float_as_uint(lo); ul += 0x7FFFu + ((ul >> 16) & 1u);
    u32 uh = __float_as_uint(hi); uh += 0x7FFFu + ((uh >> 16) & 1u);
    return (ul >> 16) | (uh & 0xFFFF0000u);
}

static __device__ __forceinline__ v8f wmma_bf16(const Frag& a, const Frag& b, v8f c) {
    return __builtin_amdgcn_wmma_f32_16x16x32_bf16(false, a.f, false, b.f, (short)0, c, false, false);
}

// A-matrix K-pattern (16-bit A 16x32, ISA 7.12.2): dword v holds K pair
static __device__ __forceinline__ int apat(int vv, int half) {
    return (vv < 4 ? vv * 2 : 16 + (vv - 4) * 2) + half * 8;
}

// Low 32 bits of a generic LDS pointer == LDS byte offset (ISA 10.2 aperture map)
static __device__ __forceinline__ u32 lds_addr(const void* p) {
    return (u32)(uintptr_t)p;
}

// CDNA5 async global->LDS copy, 16 bytes per lane (ASYNCcnt tracked)
static __device__ __forceinline__ void async_b128(u32 lds, const void* gaddr) {
    asm volatile("global_load_async_to_lds_b128 %0, %1, off"
                 :: "v"(lds), "v"(gaddr) : "memory");
}
static __device__ __forceinline__ void wait_async0() {
    asm volatile("s_wait_asynccnt 0x0" ::: "memory");
}

// ---------------------------------------------------------------------------
// Embedding + positional add -> x f32 [M,200] and padded bf16 [M,224]
// One block per token row; threads handle column pairs (packed bf16 stores).
// ---------------------------------------------------------------------------
__global__ __launch_bounds__(128) void embed_kernel(
        const int* __restrict__ tokens, const float* __restrict__ emb,
        const float* __restrict__ pos, float* __restrict__ xf,
        u16* __restrict__ xb) {
    u32 m = blockIdx.x;                  // b*S + s
    u32 s = m & (kS - 1);
    u32 t = threadIdx.x;                 // 0..127, pair index
    u32 tok = (u32)tokens[m];
    if (t < kDP / 2) {
        u32 col = t * 2;
        if (col < kD) {
            float v0 = emb[tok * kD + col]     + pos[s * kD + col];
            float v1 = emb[tok * kD + col + 1] + pos[s * kD + col + 1];
            xf[m * kD + col]     = v0;
            xf[m * kD + col + 1] = v1;
            *(u32*)&xb[m * kDP + col] = pack2bf(v0, v1);
        } else {
            *(u32*)&xb[m * kDP + col] = 0u;
        }
    }
}

// ---------------------------------------------------------------------------
// bf16 WMMA GEMM, software-pipelined with double-buffered LDS:
//   out = act(A[M,K] @ W[K,N] + bias [+ resid])
// A: bf16 (row stride lda, K-pad pre-zeroed) -> async b128 copies to LDS.
// W: f32 -> coalesced pair loads (n fast), packed bf16 dword stores to LDS.
// Block tile 128x64, 8 waves (4x2), wave tile 32x32 -> 4 WMMAs / 32-K chunk.
// Chunk c+1's A copy + B loads overlap chunk c's WMMAs.
// ---------------------------------------------------------------------------
__global__ __launch_bounds__(256) void gemm_kernel(
        const u16* __restrict__ A, u32 lda,
        const float* __restrict__ W, const float* __restrict__ bias,
        u32 N, u32 K,
        float* __restrict__ outF, u32 ldf,
        u16* __restrict__ outB, u32 ldb,
        const float* __restrict__ resid, int relu) {
    __shared__ u32 As[2][128][16];   // 2 x 8 KB, 64 B/row
    __shared__ u32 Bs[2][64][16];    // 2 x 4 KB, col-major [n][kpair]

    int tid = threadIdx.x;
    int w = tid >> 5, l = tid & 31;
    int half = l >> 4, ln = l & 15;
    int wm = w >> 1, wn = w & 1;
    u32 row0 = blockIdx.x * 128;
    u32 col0 = blockIdx.y * 64;

    v8f acc[2][2] = {{{0,0,0,0,0,0,0,0},{0,0,0,0,0,0,0,0}},
                     {{0,0,0,0,0,0,0,0},{0,0,0,0,0,0,0,0}}};

    const u32 As_base = lds_addr(&As[0][0][0]);
    const u32 nch = (K + 31) >> 5;

    // --- staging helpers -----------------------------------------------
    auto issueA = [&](u32 c, u32 buf) {
        const u8* Abase = (const u8*)(A + row0 * lda + c * 32);
        #pragma unroll
        for (int i = 0; i < 2; ++i) {
            u32 off = (u32)(tid + i * 256) * 16;   // byte offset in tile
            u32 r = off >> 6, cb = off & 63;
            async_b128(As_base + buf * 8192 + off, Abase + r * (lda * 2) + cb);
        }
    };
    // 1024 k-pairs (n fast => coalesced); clamped branch-free loads
    auto loadB = [&](u32 c, float* wv) {
        u32 k0 = c * 32;
        #pragma unroll
        for (int i = 0; i < 4; ++i) {
            u32 p = tid + i * 256;
            u32 n = col0 + (p & 63);
            u32 kk = k0 + ((p >> 6) << 1);
            u32 nc = n < N ? n : N - 1;
            u32 kc = kk < K ? kk : K - 2;    // K is even: pair valid together
            float v0 = W[kc * N + nc];
            float v1 = W[(kc + 1) * N + nc];
            bool ok = (kk < K) && (n < N);
            wv[2 * i]     = ok ? v0 : 0.0f;
            wv[2 * i + 1] = ok ? v1 : 0.0f;
        }
    };
    auto storeB = [&](const float* wv, u32 buf) {
        #pragma unroll
        for (int i = 0; i < 4; ++i) {
            u32 p = tid + i * 256;
            Bs[buf][p & 63][p >> 6] = pack2bf(wv[2 * i], wv[2 * i + 1]);
        }
    };

    // --- prologue: stage chunk 0 into buffer 0 --------------------------
    float wv[8];
    issueA(0, 0);
    loadB(0, wv);
    storeB(wv, 0);

    for (u32 c = 0; c < nch; ++c) {
        u32 buf = c & 1;
        wait_async0();        // only A(c) outstanding here -> exact wait
        __syncthreads();      // B(c) ds-stores visible, all waves ready

        float wnext[8];
        bool more = (c + 1 < nch);
        if (more) {           // overlap next chunk's copies with compute
            issueA(c + 1, buf ^ 1);
            loadB(c + 1, wnext);
        }

        Frag a[2], bfr[2];
        #pragma unroll
        for (int i = 0; i < 2; ++i) {
            int r = wm * 32 + i * 16 + ln;                 // A: M = lane%16
            #pragma unroll
            for (int vv = 0; vv < 8; ++vv)
                a[i].u[vv] = As[buf][r][vv + half * 4 + (vv >= 4 ? 4 : 0)];
        }
        #pragma unroll
        for (int j = 0; j < 2; ++j) {
            int n = wn * 32 + j * 16 + ln;                 // B: N = lane%16
            #pragma unroll
            for (int vv = 0; vv < 8; ++vv)
                bfr[j].u[vv] = Bs[buf][n][vv + half * 4 + (vv >= 4 ? 4 : 0)];  // B: K = apat(v, half)
        }
        #pragma unroll
        for (int i = 0; i < 2; ++i)
            #pragma unroll
            for (int j = 0; j < 2; ++j)
                acc[i][j] = wmma_bf16(a[i], bfr[j], acc[i][j]);

        if (more) storeB(wnext, buf ^ 1);   // after WMMAs; other buffer
        __syncthreads();                    // protect buf from next stage
    }

    // Epilogue: bias + residual + relu, f32 and/or padded-bf16 store
    #pragma unroll
    for (int i = 0; i < 2; ++i) {
        #pragma unroll
        for (int j = 0; j < 2; ++j) {
            u32 colb = col0 + wn * 32 + j * 16 + ln;
            #pragma unroll
            for (int vv = 0; vv < 8; ++vv) {
                u32 r = row0 + wm * 32 + i * 16 + vv + half * 8;  // C: M=vgpr+8*half
                if (colb < N) {
                    float vval = acc[i][j][vv] + bias[colb];
                    if (resid) vval += resid[r * N + colb];
                    if (relu)  vval = fmaxf(vval, 0.0f);
                    if (outF)  outF[r * ldf + colb] = vval;
                    if (outB)  outB[r * ldb + colb] = f2bf(vval);
                } else if (outB && colb < ldb) {
                    outB[r * ldb + colb] = 0;   // zero the K-pad columns
                }
            }
        }
    }
}

// ---------------------------------------------------------------------------
// Attention: one workgroup per (b, h, 16-q-row tile). 8 waves.
// Phase 1: stage K chunk (async b128) -> LDS; S = Q Kt via WMMA -> sc (LDS)
// Phase 2: row softmax (shfl + __expf) -> pb (LDS bf16)
// Phase 3: stage V chunk -> LDS; O = P V via WMMA -> bf16 out [M,1600]
// One 32 KB LDS union is reused as K-tile / scores / V-tile.
// ---------------------------------------------------------------------------
__global__ __launch_bounds__(256) void attn_kernel(
        const u16* __restrict__ q, const u16* __restrict__ k,
        const u16* __restrict__ v, u16* __restrict__ o) {
    __shared__ union {
        float sc[16][kS];      // scores            (32 KB)
        u32   ktu[kS][16];     // K chunk [key][16 dw] = [512][32 bf16]
        u32   vtu[32][104];    // V chunk [key][104 dw] = [32][208 bf16]
    } sh;
    __shared__ u16 pb[16][kS];  // softmax probs bf16 (16 KB)

    u32 blk = blockIdx.x;
    u32 qb = blk & 31;
    u32 h  = (blk >> 5) & 7;
    u32 b  = blk >> 8;
    int tid = threadIdx.x;
    int w = tid >> 5, l = tid & 31;
    int half = l >> 4, ln = l & 15;

    u32 headoff = b * (kS * kDH) + h * kHD;
    const u16* kb = k + headoff;
    const u16* vb = v + headoff;
    const u16* qrp = q + headoff + (qb * 16 + ln) * kDH;

    // ---- Phase 1: scores. Wave w owns key columns [w*64, w*64+64)
    v8f acc[4] = {{0,0,0,0,0,0,0,0},{0,0,0,0,0,0,0,0},
                  {0,0,0,0,0,0,0,0},{0,0,0,0,0,0,0,0}};
    const u32 kt_base = lds_addr(&sh.ktu[0][0]);
    for (int c = 0; c < 7; ++c) {
        if (c < 6) {
            // fully in-range K chunk: coalesced async b128 copies (32 KB)
            const u8* kbb = (const u8*)kb + c * 64;
            #pragma unroll
            for (int i = 0; i < 8; ++i) {
                u32 off = (u32)(tid + i * 256) * 16;     // byte in tile
                u32 key = off >> 6, cbyte = off & 63;
                async_b128(kt_base + off, kbb + key * (kDH * 2) + cbyte);
            }
            wait_async0();
        } else {
            // D-pad chunk (d in [192,224)): clamped branch-free loads
            #pragma unroll 4
            for (int i = 0; i < 32; ++i) {
                u32 e = tid + i * 256;
                u32 key = e >> 4, cp = e & 15;
                u32 d = 192 + cp * 2;
                u32 dc = d < kHD ? d : 196;
                u32 val = *(const u32*)(kb + key * kDH + dc);
                sh.ktu[key][cp] = d < kHD ? val : 0u;
            }
        }
        __syncthreads();

        Frag a;
        #pragma unroll
        for (int vv = 0; vv < 8; ++vv) {
            u32 kv = c * 32 + apat(vv, half);
            u32 kvc = kv < kHD ? kv : 196;               // clamped load
            u32 val = *(const u32*)(qrp + kvc);
            a.u[vv] = kv < kHD ? val : 0u;
        }
        #pragma unroll
        for (int j = 0; j < 4; ++j) {
            const u32* brow = sh.ktu[w * 64 + j * 16 + ln];
            Frag bfr;
            #pragma unroll
            for (int vv = 0; vv < 8; ++vv)
                bfr.u[vv] = brow[vv + half * 4 + (vv >= 4 ? 4 : 0)];
            acc[j] = wmma_bf16(a, bfr, acc[j]);
        }
        __syncthreads();
    }
    const float scale = 0.07071067811865475f;   // 1/sqrt(200)
    #pragma unroll
    for (int j = 0; j < 4; ++j)
        #pragma unroll
        for (int vv = 0; vv < 8; ++vv)
            sh.sc[vv + half * 8][w * 64 + j * 16 + ln] = acc[j][vv] * scale;
    __syncthreads();

    // ---- Phase 2: softmax. Wave w owns rows 2w, 2w+1 (16 elems/lane)
    for (int r = 2 * w; r < 2 * w + 2; ++r) {
        float mx = -1e30f;
        #pragma unroll
        for (int t = 0; t < 16; ++t) mx = fmaxf(mx, sh.sc[r][l + t * 32]);
        for (int m = 16; m > 0; m >>= 1) mx = fmaxf(mx, __shfl_xor(mx, m, 32));
        float e[16], sum = 0.0f;
        #pragma unroll
        for (int t = 0; t < 16; ++t) { e[t] = __expf(sh.sc[r][l + t * 32] - mx); sum += e[t]; }
        for (int m = 16; m > 0; m >>= 1) sum += __shfl_xor(sum, m, 32);
        float inv = 1.0f / sum;
        #pragma unroll
        for (int t = 0; t < 16; ++t) pb[r][l + t * 32] = f2bf(e[t] * inv);
    }
    __syncthreads();

    // ---- Phase 3: O = P @ V. Wave w owns out-col tiles w and w+8 (13 total)
    v8f oa0 = {0,0,0,0,0,0,0,0}, oa1 = {0,0,0,0,0,0,0,0};
    int tile1 = w + 8;
    for (int c = 0; c < 16; ++c) {
        // Stage V chunk [32 keys][208 d] (3328 dwords exactly), clamped loads
        #pragma unroll 4
        for (int i = 0; i < 13; ++i) {
            u32 e = tid + i * 256;
            u32 key = e / 104, cp = e % 104;
            u32 d = cp * 2;
            u32 dc = d < kHD ? d : 196;
            u32 val = *(const u32*)(vb + (c * 32 + key) * kDH + dc);
            sh.vtu[key][cp] = d < kHD ? val : 0u;
        }
        __syncthreads();

        Frag a;
        #pragma unroll
        for (int vv = 0; vv < 8; ++vv)
            a.u[vv] = *(const u32*)&pb[ln][c * 32 + half * 16 + vv * 2];

        const u16* vt = (const u16*)sh.vtu;   // [32][208]
        {   // tile0 = w: col = w*16+ln < 128 < 200, no guard
            u32 col = (u32)w * 16 + ln;
            Frag bfr;
            #pragma unroll
            for (int vv = 0; vv < 8; ++vv) {
                u32 kkl = half * 16 + vv * 2;
                u32 lo = vt[kkl * 208 + col], hi = vt[(kkl + 1) * 208 + col];
                bfr.u[vv] = lo | (hi << 16);
            }
            oa0 = wmma_bf16(a, bfr, oa0);
        }
        if (tile1 < 13) {
            u32 col = (u32)tile1 * 16 + ln;
            Frag bfr;
            #pragma unroll
            for (int vv = 0; vv < 8; ++vv) {
                u32 kkl = half * 16 + vv * 2;
                u32 lo = vt[kkl * 208 + col], hi = vt[(kkl + 1) * 208 + col];
                bfr.u[vv] = lo | (hi << 16);
            }
            oa1 = wmma_bf16(a, bfr, oa1);
        }
        __syncthreads();
    }

    // Store output (bf16, [M, 1600])
    u32 obase = headoff + (qb * 16 + half * 8) * kDH;
    #pragma unroll
    for (int vv = 0; vv < 8; ++vv)
        o[obase + vv * kDH + w * 16 + ln] = f2bf(oa0[vv]);
    if (tile1 < 13) {
        u32 col = (u32)tile1 * 16 + ln;
        if (col < kHD)
            #pragma unroll
            for (int vv = 0; vv < 8; ++vv)
                o[obase + vv * kDH + col] = f2bf(oa1[vv]);
    }
}

// ---------------------------------------------------------------------------
// Joint LayerNorm over (S, D) per batch. One 1024-thread block per batch.
// Normalize loop works on column pairs -> packed bf16 dword stores.
// ---------------------------------------------------------------------------
__global__ __launch_bounds__(1024) void ln_kernel(const float* __restrict__ in,
                                                  float* __restrict__ outF,
                                                  u16* __restrict__ outB) {
    __shared__ float red[64];
    u32 b = blockIdx.x;
    const float* src = in + b * (kS * kD);
    float s = 0.0f, s2 = 0.0f;
    for (u32 i = threadIdx.x; i < kS * kD; i += 1024) {
        float x = src[i]; s += x; s2 += x * x;
    }
    for (int m = 16; m > 0; m >>= 1) { s += __shfl_xor(s, m, 32); s2 += __shfl_xor(s2, m, 32); }
    int wid = threadIdx.x >> 5;
    if ((threadIdx.x & 31) == 0) { red[wid] = s; red[32 + wid] = s2; }
    __syncthreads();
    if (threadIdx.x < 32) {
        float a = red[threadIdx.x], c = red[32 + threadIdx.x];
        for (int m = 16; m > 0; m >>= 1) { a += __shfl_xor(a, m, 32); c += __shfl_xor(c, m, 32); }
        if (threadIdx.x == 0) { red[0] = a; red[1] = c; }
    }
    __syncthreads();
    const float invn = 1.0f / (float)(kS * kD);
    float mean = red[0] * invn;
    float var  = red[1] * invn - mean * mean;
    float rs = rsqrtf(var + 1e-5f);
    for (u32 pi = threadIdx.x; pi < kS * (kDP / 2); pi += 1024) {
        u32 srow = pi / (kDP / 2), col = (pi % (kDP / 2)) * 2;
        u32 m = b * kS + srow;
        if (col < kD) {
            float x0 = (src[srow * kD + col]     - mean) * rs;
            float x1 = (src[srow * kD + col + 1] - mean) * rs;
            outF[m * kD + col]     = x0;
            outF[m * kD + col + 1] = x1;
            *(u32*)&outB[m * kDP + col] = pack2bf(x0, x1);
        } else {
            *(u32*)&outB[m * kDP + col] = 0u;
        }
    }
}

// ---------------------------------------------------------------------------
// Classifier: out[b,c] = x[b,0,:] @ Wc + bc   (640 outputs)
// ---------------------------------------------------------------------------
__global__ void cls_kernel(const float* __restrict__ x, const float* __restrict__ Wc,
                           const float* __restrict__ bc, float* __restrict__ out) {
    int i = blockIdx.x * blockDim.x + threadIdx.x;
    if (i >= kB * kC) return;
    u32 b = i / kC, c = i % kC;
    const float* xr = x + b * (kS * kD);    // token 0 of batch b
    float s = bc[c];
    for (int d = 0; d < kD; ++d) s += xr[d] * Wc[(u32)d * kC + c];
    out[i] = s;
}

// ---------------------------------------------------------------------------
extern "C" void kernel_launch(void* const* d_in, const int* in_sizes, int n_in,
                              void* d_out, int out_size, void* d_ws, size_t ws_size,
                              hipStream_t stream) {
    const int*   tokens = (const int*)  d_in[0];
    const float* emb    = (const float*)d_in[1];
    const float* pos    = (const float*)d_in[2];
    const float* Wq = (const float*)d_in[3];  const float* bq = (const float*)d_in[4];
    const float* Wk = (const float*)d_in[5];  const float* bk = (const float*)d_in[6];
    const float* Wv = (const float*)d_in[7];  const float* bv = (const float*)d_in[8];
    const float* Wf = (const float*)d_in[9];  const float* bf = (const float*)d_in[10];
    const float* W1 = (const float*)d_in[11]; const float* b1 = (const float*)d_in[12];
    const float* W2 = (const float*)d_in[13]; const float* b2 = (const float*)d_in[14];
    const float* Wc = (const float*)d_in[15]; const float* bc = (const float*)d_in[16];
    float* out = (float*)d_out;

    // Workspace carve
    char* p = (char*)d_ws;
    auto carve = [&](size_t bytes) { char* r = p; p += (bytes + 255) & ~(size_t)255; return r; };
    float* xf  = (float*)carve((size_t)kM * kD * 4);     // master f32 activations
    u16*   xb  = (u16*)  carve((size_t)kM * kDP * 2);    // padded bf16 shadow
    float* res = (float*)carve((size_t)kM * kD * 4);     // pre-LN residual sum
    u16*   qb_ = (u16*)  carve((size_t)kM * kDH * 2);
    u16*   kb_ = (u16*)  carve((size_t)kM * kDH * 2);
    u16*   vb_ = (u16*)  carve((size_t)kM * kDH * 2);
    u16*   cb_ = (u16*)  carve((size_t)kM * kDH * 2);    // attention output
    u16*   hb_ = (u16*)  carve((size_t)kM * kDP * 2);    // FFN hidden (padded)

    dim3 blk256(256);

    embed_kernel<<<kM, dim3(128), 0, stream>>>(tokens, emb, pos, xf, xb);

    dim3 gQKV(kM / 128, (kDH + 63) / 64);   // 256 x 25
    dim3 gD  (kM / 128, (kD  + 63) / 64);   // 256 x 4
    dim3 gAtt(kB * kH * (kS / 16));         // 16384

    for (int l = 0; l < kL; ++l) {
        const float* wq = Wq + (size_t)l * kD * kDH;  const float* bq_ = bq + (size_t)l * kDH;
        const float* wk = Wk + (size_t)l * kD * kDH;  const float* bk_ = bk + (size_t)l * kDH;
        const float* wv = Wv + (size_t)l * kD * kDH;  const float* bv_ = bv + (size_t)l * kDH;
        const float* wf = Wf + (size_t)l * kDH * kD;  const float* bf_ = bf + (size_t)l * kD;
        const float* w1 = W1 + (size_t)l * kD * kD;   const float* b1_ = b1 + (size_t)l * kD;
        const float* w2 = W2 + (size_t)l * kD * kD;   const float* b2_ = b2 + (size_t)l * kD;

        gemm_kernel<<<gQKV, blk256, 0, stream>>>(xb, kDP, wq, bq_, kDH, kD,
                                                 nullptr, 0, qb_, kDH, nullptr, 0);
        gemm_kernel<<<gQKV, blk256, 0, stream>>>(xb, kDP, wk, bk_, kDH, kD,
                                                 nullptr, 0, kb_, kDH, nullptr, 0);
        gemm_kernel<<<gQKV, blk256, 0, stream>>>(xb, kDP, wv, bv_, kDH, kD,
                                                 nullptr, 0, vb_, kDH, nullptr, 0);
        attn_kernel<<<gAtt, blk256, 0, stream>>>(qb_, kb_, vb_, cb_);
        gemm_kernel<<<gD, blk256, 0, stream>>>(cb_, kDH, wf, bf_, kD, kDH,
                                               res, kD, nullptr, 0, xf, 0);
        ln_kernel<<<kB, 1024, 0, stream>>>(res, xf, xb);
        gemm_kernel<<<gD, blk256, 0, stream>>>(xb, kDP, w1, b1_, kD, kD,
                                               nullptr, 0, hb_, kDP, nullptr, 1);
        gemm_kernel<<<gD, blk256, 0, stream>>>(hb_, kDP, w2, b2_, kD, kD,
                                               res, kD, nullptr, 0, xf, 0);
        ln_kernel<<<kB, 1024, 0, stream>>>(res, xf, xb);
    }

    cls_kernel<<<(kB * kC + 255) / 256, blk256, 0, stream>>>(xf, Wc, bc, out);
    (void)in_sizes; (void)n_in; (void)out_size; (void)ws_size;
}
